// AlibiMultiHeadSelfAttention_9062380994610
// MI455X (gfx1250) — matrix operand order repaired and checked by reading, not compiled, not run on hardware
//
#include <hip/hip_runtime.h>
#include <hip/hip_bf16.h>
#include <math.h>

// ---------------------------------------------------------------------------
// ALiBi multi-head self-attention for ViT on MI455X (gfx1250, wave32, WMMA).
// B=16, S=577, D=768, H=12, hd=64.  d_out = [out f32 | attention_weights f32].
// Matmuls: v_wmma_f32_16x16x32_bf16.  Attention core is fused per 16-query
// tile (logits+ALiBi+softmax+P@V in LDS); K tiles are staged into LDS by the
// Tensor Data Mover (tensor_load_to_lds), double-buffered against compute.
// ---------------------------------------------------------------------------

typedef __attribute__((ext_vector_type(16))) __bf16 v16bf;
typedef __attribute__((ext_vector_type(8)))  __bf16 v8bf;
typedef __attribute__((ext_vector_type(4)))  __bf16 v4bf;
typedef __attribute__((ext_vector_type(8)))  float  v8f;
typedef __attribute__((ext_vector_type(4)))  float  v4f;
typedef __attribute__((ext_vector_type(4)))  unsigned int u32x4;
typedef __attribute__((ext_vector_type(8)))  int    i32x8;
typedef __attribute__((ext_vector_type(4)))  int    i32x4;

#define Bn   16
#define Sn   577
#define Dn   768
#define Hn   12
#define HDn  64
#define SPn  640            // padded S: 10 * 64, covers all 16-row / 64-col tiles
#define BSn  (Bn * Sn)      // 9232 = 577 * 16  -> M tiles need no bounds checks

#define LFS  644            // LDS f32 logits row stride (bank-skewed)
#define PBS  648            // LDS bf16 prob row stride (16B aligned + bank-spread)
#define KSS  72             // LDS bf16 K-tile row stride (144B: aligned + skewed)

// ---------------------------------------------------------------------------
// WMMA fragment helpers (contiguous k = 16h + s order, same in A and B)
// ---------------------------------------------------------------------------
__device__ __forceinline__ int lane_id() { return (int)(threadIdx.x & 31); }

// A fragment: 16(M) x 32(K) bf16, source row-major with `stride` elements/row.
// Lane L<16: M=L, K = 0..15; lane L>=16: M=L-16, K = 16..31.
__device__ __forceinline__ v16bf load_afrag_bf16(const __bf16* tile, int stride) {
    int l  = lane_id();
    int m  = l & 15;
    int k0 = (l & 16) ? 16 : 0;
    const __bf16* p = tile + (size_t)m * stride + k0;
    v16bf f;
    *(v8bf*)&f       = *(const v8bf*)(p);       // K = k0 .. k0+7
    *((v8bf*)&f + 1) = *(const v8bf*)(p + 8);   // K = k0+8 .. k0+15
    return f;
}

// B fragment: 32(K) x 16(N) bf16.  Source is the *transposed* matrix, i.e.
// row-major [N][K] with `stride` elements/row (so each lane reads one N row).
// Lane L<16: N=L, K=0..15; lane L>=16: N=L-16, K=16..31.
__device__ __forceinline__ v16bf load_bfrag_bf16(const __bf16* nmat, int stride) {
    int l  = lane_id();
    int n  = l & 15;
    int k0 = (l & 16) ? 16 : 0;
    const __bf16* p = nmat + (size_t)n * stride + k0;
    v16bf f;
    *(v8bf*)&f       = *(const v8bf*)(p);
    *((v8bf*)&f + 1) = *(const v8bf*)(p + 8);
    return f;
}

__device__ __forceinline__ v8f wmma_bf16(v16bf a, v16bf b, v8f c) {
    return __builtin_amdgcn_wmma_f32_16x16x32_bf16(
        /*neg_a=*/false, a, /*neg_b=*/false, b,
        /*c_mod=*/(short)0, c, /*reuse_a=*/false, /*reuse_b=*/false);
}

// ---------------------------------------------------------------------------
// TDM: DMA one 64x64 bf16 K tile (global, row stride 64 elems, contiguous)
// into LDS with a padded destination stride of KSS (=72) elems.
// D# fields per CDNA5 ISA 8.3/8.4:
//   g0: count=1 | lds_addr | global_addr[56:0] | type=2
//   g1: data_size=2B, pad_enable, pad_interval=32 DW (128B), pad_amount=4 DW
//       tensor_dim0=64, tensor_dim1=64, tile_dim0=64, tile_dim1=64,
//       tensor_dim0_stride=64
// ---------------------------------------------------------------------------
__device__ __forceinline__ void tdm_load_k_tile(const __bf16* gsrc,
                                                unsigned int lds_addr) {
    unsigned long long ga = (unsigned long long)(size_t)gsrc;
    u32x4 g0 = { 1u,                                   // count=1 (valid D#)
                 lds_addr,                             // lds_addr (bytes)
                 (unsigned int)ga,                     // global_addr[31:0]
                 (unsigned int)((ga >> 32) & 0x01FFFFFFull) | (2u << 30) };
    i32x8 g1 = { (1 << 16) | (1 << 20) | (4 << 22) | (3 << 25),
                 (64 << 16),        // tensor_dim0 = 64 (bits[63:48])
                 (64 << 16),        // tensor_dim1 = 64 (bits[95:80])
                 (64 << 16),        // tile_dim0   = 64 (bits[127:112])
                 64,                // tile_dim1   = 64 (bits[143:128])
                 64,                // tensor_dim0_stride = 64 (bits[191:160])
                 0, 0 };
    i32x4 gz = { 0, 0, 0, 0 };
#if __clang_major__ >= 23
    i32x8 gz8 = { 0, 0, 0, 0, 0, 0, 0, 0 };
    __builtin_amdgcn_tensor_load_to_lds(g0, g1, gz, gz, gz8, 0);
#else
    __builtin_amdgcn_tensor_load_to_lds(g0, g1, gz, gz, 0);
#endif
}

// ---------------------------------------------------------------------------
// K0: f32 -> bf16 bulk convert (n4 = element count / 4)
// ---------------------------------------------------------------------------
__global__ __launch_bounds__(256) void cvt_f32_to_bf16(
    const float* __restrict__ src, __bf16* __restrict__ dst, int n4) {
    int i = blockIdx.x * blockDim.x + threadIdx.x;
    if (i < n4) {
        v4f v = ((const v4f*)src)[i];
        v4bf o;
        o.x = (__bf16)v.x; o.y = (__bf16)v.y;
        o.z = (__bf16)v.z; o.w = (__bf16)v.w;
        ((v4bf*)dst)[i] = o;
    }
}

// Zero-fill bf16 buffer (n4 = element count / 4).  Needed so V's padded
// columns are true zeros (P pad is zero; 0 * 0 keeps padded K-range inert).
__global__ __launch_bounds__(256) void zero_bf16(__bf16* __restrict__ dst, int n4) {
    int i = blockIdx.x * blockDim.x + threadIdx.x;
    if (i < n4) {
        v4bf z = { (__bf16)0.0f, (__bf16)0.0f, (__bf16)0.0f, (__bf16)0.0f };
        ((v4bf*)dst)[i] = z;
    }
}

// ---------------------------------------------------------------------------
// K1: QKV projection.  out[r,e] = sum_d X[r,d] * W[e,d]  (torch Linear).
// Grid: x = M tile (577), y = head (12, N tile of 64), z = mode (0=Q,1=K,2=V).
// Q,K stored [B,H,SPn,64]; V stored transposed [B,H,64,SPn].
// ---------------------------------------------------------------------------
__global__ __launch_bounds__(32) void qkv_gemm(
    const __bf16* __restrict__ X, const __bf16* __restrict__ W,
    __bf16* __restrict__ Qb, __bf16* __restrict__ Kb, __bf16* __restrict__ Vt) {
    int mt = blockIdx.x, nt = blockIdx.y, mode = blockIdx.z;
    const __bf16* Wsel = W + (size_t)mode * Dn * Dn;
    int r0 = mt * 16;
    int e0 = nt * 64;
    v8f acc[4] = {};
    for (int k = 0; k < Dn; k += 32) {
        v16bf a = load_afrag_bf16(X + (size_t)r0 * Dn + k, Dn);
#pragma unroll
        for (int j = 0; j < 4; ++j) {
            v16bf bf = load_bfrag_bf16(Wsel + (size_t)(e0 + j * 16) * Dn + k, Dn);
            acc[j] = wmma_bf16(a, bf, acc[j]);
        }
    }
    int l = lane_id(), n = l & 15, hi = (l >> 4) & 1;
#pragma unroll
    for (int j = 0; j < 4; ++j) {
        int d = j * 16 + n;
#pragma unroll
        for (int i = 0; i < 8; ++i) {
            int r = r0 + i + hi * 8;      // always < 9232
            int b = r / Sn;
            int s = r - b * Sn;
            size_t bh = (size_t)b * Hn + nt;
            __bf16 v = (__bf16)acc[j][i];
            if (mode == 2)      Vt[(bh * HDn + d) * SPn + s] = v;
            else if (mode == 1) Kb[(bh * SPn + s) * HDn + d] = v;
            else                Qb[(bh * SPn + s) * HDn + d] = v;
        }
    }
}

// ---------------------------------------------------------------------------
// K2: fused attention row-block:
//   TDM-staged K tiles -> logits(QK^T/8 + alibi) in LDS -> softmax (shfl) ->
//   attention_weights (one HBM write) -> P@V from LDS -> wv (bf16).
// Grid: x = q tile (37), y = b*H (192).  Block: 128 threads = 4 wave32s.
// ---------------------------------------------------------------------------
__global__ __launch_bounds__(128) void fused_attn(
    const __bf16* __restrict__ Qb, const __bf16* __restrict__ Kb,
    const __bf16* __restrict__ Vt, float* __restrict__ attnW,
    __bf16* __restrict__ wv) {
    int qt = blockIdx.x, bh = blockIdx.y;
    int b = bh / Hn, h = bh % Hn;
    int q0 = qt * 16;
    int tid = (int)threadIdx.x;
    int w = tid >> 5;
    int l = tid & 31, n = l & 15, hi = (l >> 4) & 1;

    __shared__ float  Lf[16][LFS];               // logits / probs, f32
    __shared__ __bf16 Pbf[16][PBS];              // probs bf16, zero-padded
    __shared__ __align__(16) __bf16 Ks[2][64][KSS];  // TDM-staged K tiles

    const __bf16* Qbase  = Qb + ((size_t)bh * SPn + q0) * HDn;
    const __bf16* Kslice = Kb + (size_t)bh * SPn * HDn;
    float slope = (h < 8) ? exp2f(-(float)(h + 1))
                          : exp2f(-((float)(h - 8) + 0.5f));

    // ---- Phase 1: logits; K tiles DMA'd by TDM, double-buffered ------------
    v16bf aq0 = load_afrag_bf16(Qbase + 0,  HDn);   // loop-invariant Q frags
    v16bf aq1 = load_afrag_bf16(Qbase + 32, HDn);
    if (w == 0)
        tdm_load_k_tile(Kslice, (unsigned int)(size_t)&Ks[0][0][0]);

    for (int kt = 0; kt < SPn / 64; ++kt) {
        if (w == 0) __builtin_amdgcn_s_wait_tensorcnt(0);
        __syncthreads();          // publish buffer kt; all waves done with kt-1
        if (w == 0 && kt + 1 < SPn / 64)
            tdm_load_k_tile(Kslice + (size_t)(kt + 1) * 64 * HDn,
                            (unsigned int)(size_t)&Ks[(kt + 1) & 1][0][0]);
        int buf = kt & 1;
        v8f acc = {};
        acc = wmma_bf16(aq0, load_bfrag_bf16(&Ks[buf][w * 16][0],  KSS), acc);
        acc = wmma_bf16(aq1, load_bfrag_bf16(&Ks[buf][w * 16][32], KSS), acc);

        int ki = kt * 64 + w * 16 + n;
#pragma unroll
        for (int i = 0; i < 8; ++i) {
            int m = i + hi * 8;
            int qi = q0 + m;
            float bias = 0.0f;
            if (qi > 0 && qi < Sn && ki > 0 && ki < Sn) {
                int p = qi - 1, q2 = ki - 1;
                int dy = p / 24 - q2 / 24;
                int dx = p % 24 - q2 % 24;
                bias = -slope * sqrtf((float)(dy * dy + dx * dx));
            }
            Lf[m][ki] = acc[i] * 0.125f + bias;
        }
    }
    __syncthreads();

    // ---- Phase 2: softmax, 8 lanes per row, shfl_xor reductions ------------
    {
        int row = tid >> 3, c0 = tid & 7;
        float mx = -3.4e38f;
        for (int c = c0; c < Sn; c += 8) mx = fmaxf(mx, Lf[row][c]);
#pragma unroll
        for (int d = 1; d < 8; d <<= 1) mx = fmaxf(mx, __shfl_xor(mx, d, 32));
        float sum = 0.0f;
        for (int c = c0; c < Sn; c += 8) {
            float e = __expf(Lf[row][c] - mx);
            Lf[row][c] = e;
            sum += e;
        }
#pragma unroll
        for (int d = 1; d < 8; d <<= 1) sum += __shfl_xor(sum, d, 32);
        float inv = 1.0f / sum;
        for (int c = c0; c < SPn; c += 8) {
            float p = (c < Sn) ? Lf[row][c] * inv : 0.0f;
            if (c < Sn) Lf[row][c] = p;
            Pbf[row][c] = (__bf16)p;     // pad cols forced to zero
        }
    }
    __syncthreads();

    // ---- Phase 3: single coalesced HBM write of attention weights ----------
    for (int r2 = 0; r2 < 16; ++r2) {
        int qi = q0 + r2;
        if (qi >= Sn) break;
        float* dst = attnW + ((size_t)bh * Sn + qi) * Sn;
        for (int c = tid; c < Sn; c += 128) dst[c] = Lf[r2][c];
    }

    // ---- Phase 4: wv = P @ V ; wave w owns d-tile [w*16, w*16+16) ----------
    v8f acc = {};
    for (int kb = 0; kb < SPn; kb += 32) {
        v16bf a  = load_afrag_bf16(&Pbf[0][kb], PBS);          // ds_load_b128
        v16bf bf = load_bfrag_bf16(
            Vt + ((size_t)bh * HDn + w * 16) * SPn + kb, SPn);
        acc = wmma_bf16(a, bf, acc);
    }
    int d = w * 16 + n;
#pragma unroll
    for (int i = 0; i < 8; ++i) {
        int qi = q0 + i + hi * 8;
        if (qi < Sn)
            wv[((size_t)b * Sn + qi) * Dn + h * HDn + d] = (__bf16)acc[i];
    }
}

// ---------------------------------------------------------------------------
// K3: out = wv @ Wo^T + bo  -> f32 out region.  Grid: x=577 M tiles, y=12.
// ---------------------------------------------------------------------------
__global__ __launch_bounds__(32) void out_gemm(
    const __bf16* __restrict__ WV, const __bf16* __restrict__ Wo,
    const float* __restrict__ bo, float* __restrict__ out) {
    int mt = blockIdx.x, nt = blockIdx.y;
    int r0 = mt * 16, e0 = nt * 64;
    v8f acc[4] = {};
    for (int k = 0; k < Dn; k += 32) {
        v16bf a = load_afrag_bf16(WV + (size_t)r0 * Dn + k, Dn);
#pragma unroll
        for (int j = 0; j < 4; ++j) {
            v16bf bf = load_bfrag_bf16(Wo + (size_t)(e0 + j * 16) * Dn + k, Dn);
            acc[j] = wmma_bf16(a, bf, acc[j]);
        }
    }
    int l = lane_id(), n = l & 15, hi = (l >> 4) & 1;
#pragma unroll
    for (int j = 0; j < 4; ++j) {
        int e = e0 + j * 16 + n;
        float bias = bo[e];
#pragma unroll
        for (int i = 0; i < 8; ++i) {
            int r = r0 + i + hi * 8;
            out[(size_t)r * Dn + e] = acc[j][i] + bias;
        }
    }
}

// ---------------------------------------------------------------------------
// Launch
// ---------------------------------------------------------------------------
extern "C" void kernel_launch(void* const* d_in, const int* in_sizes, int n_in,
                              void* d_out, int out_size, void* d_ws, size_t ws_size,
                              hipStream_t stream) {
    const float* X  = (const float*)d_in[0];
    const float* Wq = (const float*)d_in[1];
    const float* Wk = (const float*)d_in[2];
    const float* Wv = (const float*)d_in[3];
    const float* Wo = (const float*)d_in[4];
    const float* bo = (const float*)d_in[5];

    float* out  = (float*)d_out;
    float* attn = out + (size_t)BSn * Dn;   // attention_weights region

    // Workspace layout (bf16), ~80 MB total
    char* ws = (char*)d_ws;
    __bf16* Xbf = (__bf16*)ws;  ws += (size_t)BSn * Dn * 2;
    __bf16* Wbf = (__bf16*)ws;  ws += (size_t)4 * Dn * Dn * 2;     // Wq,Wk,Wv,Wo
    __bf16* Qb  = (__bf16*)ws;  ws += (size_t)Bn * Hn * SPn * HDn * 2;
    __bf16* Kb  = (__bf16*)ws;  ws += (size_t)Bn * Hn * SPn * HDn * 2;
    __bf16* Vt  = (__bf16*)ws;  ws += (size_t)Bn * Hn * HDn * SPn * 2;
    __bf16* WVb = (__bf16*)ws;  ws += (size_t)BSn * Dn * 2;

    const int DD4 = Dn * Dn / 4;
    const int VT4 = Bn * Hn * HDn * SPn / 4;
    cvt_f32_to_bf16<<<(BSn * Dn / 4 + 255) / 256, 256, 0, stream>>>(X, Xbf, BSn * Dn / 4);
    cvt_f32_to_bf16<<<(DD4 + 255) / 256, 256, 0, stream>>>(Wq, Wbf + 0 * (size_t)Dn * Dn, DD4);
    cvt_f32_to_bf16<<<(DD4 + 255) / 256, 256, 0, stream>>>(Wk, Wbf + 1 * (size_t)Dn * Dn, DD4);
    cvt_f32_to_bf16<<<(DD4 + 255) / 256, 256, 0, stream>>>(Wv, Wbf + 2 * (size_t)Dn * Dn, DD4);
    cvt_f32_to_bf16<<<(DD4 + 255) / 256, 256, 0, stream>>>(Wo, Wbf + 3 * (size_t)Dn * Dn, DD4);
    zero_bf16<<<(VT4 + 255) / 256, 256, 0, stream>>>(Vt, VT4);

    qkv_gemm<<<dim3(BSn / 16, Hn, 3), 32, 0, stream>>>(Xbf, Wbf, Qb, Kb, Vt);
    fused_attn<<<dim3(37, Bn * Hn), 128, 0, stream>>>(Qb, Kb, Vt, attn, WVb);
    out_gemm<<<dim3(BSn / 16, Hn), 32, 0, stream>>>(WVb, Wbf + 3 * (size_t)Dn * Dn, bo, out);
}
